// Net_11914239279185
// MI455X (gfx1250) — matrix operand order repaired and checked by reading, not compiled, not run on hardware
//
#include <hip/hip_runtime.h>
#include <hip/hip_fp16.h>
#include <math.h>

typedef __attribute__((ext_vector_type(16))) _Float16 v16h;
typedef __attribute__((ext_vector_type(8)))  float    v8f;

#define NN    50000
#define NE    1600000
#define FIN   128
#define HG    64      // GRU hidden
#define G3    192     // 3*H
#define NCLS  10

union V16H { v16h v; _Float16 e[16]; };
union V8F  { v8f  v; float    e[8];  };

// ---------------- degree / norm ----------------
__global__ void k_degree(const int* __restrict__ col, float* __restrict__ deg,
                         int E, int n) {
    int i = blockIdx.x * blockDim.x + threadIdx.x;
    if (i >= E + n) return;
    int c = (i < E) ? col[i] : (i - E);     // self-loops appended
    atomicAdd(&deg[c], 1.0f);
}

__global__ void k_dinv(float* __restrict__ deg, int n) {
    int i = blockIdx.x * blockDim.x + threadIdx.x;
    if (i < n) deg[i] = rsqrtf(fmaxf(deg[i], 1e-12f));
}

// zero-pad W2 [16x32] -> [32x32] so conv2 runs guard-free at K=32
__global__ void k_pad_w2(const float* __restrict__ w2, float* __restrict__ wp) {
    int i = blockIdx.x * blockDim.x + threadIdx.x;
    if (i >= 32 * 32) return;
    wp[i] = ((i >> 5) < 16) ? w2[i] : 0.0f;
}

// ---------------- WMMA GEMM: C[M x N] = A[M x K](f32) * B + bias ----------------
// One wave computes a 16-row stripe and NT consecutive 16-col tiles, keeping all
// A K-chunks in registers. A and TRANS_B=true B fragment loads are contiguous float4 (b128).
// TRANS_B=false: B is [K x N] row-major (ldb = N).   (GCN weights, ldb arbitrary)
// TRANS_B=true : B is [N x K] row-major (ldb = K).   (GRU wih: gx = h @ wih^T)
template <int K, bool TRANS_B, int NT>
__global__ void k_gemm_wmma(const float* __restrict__ A, int lda,
                            const float* __restrict__ B, int ldb,
                            const float* __restrict__ bias,
                            float* __restrict__ C, int ldc) {
    constexpr int KC = (K + 31) / 32;
    const int lane   = threadIdx.x;               // block = 32 (one wave)
    const int m0     = blockIdx.x * 16;
    const int n_base = blockIdx.y * (NT * 16);
    const int nl  = lane & 15;
    const int kbA = (lane < 16) ? 0 : 16;         // A: 16x32, K halves by lane half
    const int kbB = (lane < 16) ? 0 : 16;         // B: 32x16, K halves by lane half
    const int mA  = m0 + nl;

    // ---- load & convert all A fragments once (2x b128 per 16-K group) ----
    const float4* __restrict__ arow4 = (const float4*)(A + (size_t)mA * lda);
    V16H a[KC];
    #pragma unroll
    for (int kc = 0; kc < KC; ++kc) {
        #pragma unroll
        for (int g = 0; g < 2; ++g) {
            float4 p0 = {0.f, 0.f, 0.f, 0.f}, p1 = {0.f, 0.f, 0.f, 0.f};
            if (kc * 32 + g * 16 < K) {           // constant-folds after unroll
                p0 = arow4[kc * 8 + g * 2 + (kbA >> 2)];
                p1 = arow4[kc * 8 + g * 2 + (kbA >> 2) + 1];
            }
            a[kc].e[g * 8 + 0] = (_Float16)p0.x;
            a[kc].e[g * 8 + 1] = (_Float16)p0.y;
            a[kc].e[g * 8 + 2] = (_Float16)p0.z;
            a[kc].e[g * 8 + 3] = (_Float16)p0.w;
            a[kc].e[g * 8 + 4] = (_Float16)p1.x;
            a[kc].e[g * 8 + 5] = (_Float16)p1.y;
            a[kc].e[g * 8 + 6] = (_Float16)p1.z;
            a[kc].e[g * 8 + 7] = (_Float16)p1.w;
        }
    }

    const int mr = m0 + ((lane >= 16) ? 8 : 0);

    #pragma unroll
    for (int nt = 0; nt < NT; ++nt) {
        const int n = n_base + nt * 16 + nl;
        V8F c;
        #pragma unroll
        for (int r = 0; r < 8; ++r) c.e[r] = 0.0f;

        #pragma unroll
        for (int kc = 0; kc < KC; ++kc) {
            V16H b;
            if (TRANS_B) {                        // contiguous 16 floats -> 4x b128
                const float4* __restrict__ brow4 = (const float4*)(B + (size_t)n * ldb);
                #pragma unroll
                for (int q = 0; q < 4; ++q) {
                    float4 p = brow4[kc * 8 + (kbB >> 2) + q];
                    b.e[q * 4 + 0] = (_Float16)p.x;
                    b.e[q * 4 + 1] = (_Float16)p.y;
                    b.e[q * 4 + 2] = (_Float16)p.z;
                    b.e[q * 4 + 3] = (_Float16)p.w;
                }
            } else {                              // small weight matrix, strided
                #pragma unroll
                for (int j = 0; j < 16; ++j) {
                    int k = kc * 32 + kbB + j;
                    b.e[j] = (_Float16)B[(size_t)k * ldb + n];
                }
            }
            c.v = __builtin_amdgcn_wmma_f32_16x16x32_f16(
                      false, a[kc].v, false, b.v, (short)0, c.v, false, false);
        }

        const float bb = bias ? bias[n] : 0.0f;
        #pragma unroll
        for (int r = 0; r < 8; ++r)
            C[(size_t)(mr + r) * ldc + n] = c.e[r] + bb;
    }
}

// ---------------- normalized scatter-add over edges ----------------
template <int F>
__global__ void k_scatter(const int* __restrict__ ei, const float* __restrict__ dinv,
                          const float* __restrict__ xt, float* __restrict__ acc,
                          int E, int n) {
    int i = blockIdx.x * blockDim.x + threadIdx.x;
    if (i >= E + n) return;
    int r, c;
    if (i < E) { r = ei[i]; c = ei[E + i]; } else { r = c = i - E; }
    const float norm = dinv[r] * dinv[c];
    const float4* __restrict__ src = (const float4*)(xt + (size_t)r * F);
    float* __restrict__ dst = acc + (size_t)c * F;
    #pragma unroll
    for (int q = 0; q < F / 4; ++q) {
        float4 v = src[q];
        atomicAdd(&dst[q * 4 + 0], v.x * norm);
        atomicAdd(&dst[q * 4 + 1], v.y * norm);
        atomicAdd(&dst[q * 4 + 2], v.z * norm);
        atomicAdd(&dst[q * 4 + 3], v.w * norm);
    }
}

__global__ void k_bias_relu(float* __restrict__ h, const float* __restrict__ b,
                            int n, int F) {
    int i = blockIdx.x * blockDim.x + threadIdx.x;
    if (i >= n * F) return;
    float v = h[i] + b[i % F];
    h[i] = v > 0.0f ? v : 0.0f;
}

// ---------------- serial bidirectional GRU scan ----------------
// grid = 2 (dir 0 = forward, dir 1 = backward), block = 192 threads (6 waves).
// Whh (48KB) stays resident in LDS; per step: 64-FMA dot per thread + 2 barriers.
__global__ void k_gru_scan(const float* __restrict__ gx_f, const float* __restrict__ gx_b,
                           const float* __restrict__ whh_f, const float* __restrict__ bhh_f,
                           const float* __restrict__ whh_b, const float* __restrict__ bhh_b,
                           float* __restrict__ y, int T) {
    __shared__ float w_s[G3 * HG];
    __shared__ float h_s[HG];
    __shared__ float r_s[HG], z_s[HG], gn_s[HG], ghn_s[HG];

    const int dir = blockIdx.x;
    const float* __restrict__ gx  = dir ? gx_b  : gx_f;
    const float* __restrict__ whh = dir ? whh_b : whh_f;
    const float* __restrict__ bhh = dir ? bhh_b : bhh_f;
    const int tid = threadIdx.x;                  // 0..191

    for (int i = tid; i < G3 * HG; i += blockDim.x) w_s[i] = whh[i];
    if (tid < HG) h_s[tid] = 0.0f;
    __syncthreads();

    const float bh = bhh[tid];
    const float* __restrict__ wrow = &w_s[tid * HG];

    for (int s = 0; s < T; ++s) {
        const int t = dir ? (T - 1 - s) : s;
        float gh = bh;
        #pragma unroll 8
        for (int k = 0; k < HG; ++k) gh += wrow[k] * h_s[k];
        const float g = gx[(size_t)t * G3 + tid];
        if (tid < HG)             r_s[tid]          = 1.0f / (1.0f + __expf(-(g + gh)));
        else if (tid < 2 * HG)    z_s[tid - HG]     = 1.0f / (1.0f + __expf(-(g + gh)));
        else { gn_s[tid - 2*HG] = g; ghn_s[tid - 2*HG] = gh; }
        __syncthreads();
        if (tid < HG) {
            const float nn = tanhf(gn_s[tid] + r_s[tid] * ghn_s[tid]);
            const float z  = z_s[tid];
            const float hn = (1.0f - z) * nn + z * h_s[tid];
            h_s[tid] = hn;
            y[(size_t)t * (2 * HG) + dir * HG + tid] = hn;   // concat [fwd | bwd]
        }
        __syncthreads();
    }
}

// ---------------- final linear head ----------------
__global__ void k_linear(const float* __restrict__ y, const float* __restrict__ w,
                         const float* __restrict__ b, float* __restrict__ out, int n) {
    int i = blockIdx.x * blockDim.x + threadIdx.x;
    if (i >= n * NCLS) return;
    const int node = i / NCLS, c = i % NCLS;
    const float4* __restrict__ yr = (const float4*)(y + (size_t)node * (2 * HG));
    const float4* __restrict__ wr = (const float4*)(w + (size_t)c * (2 * HG));
    float s = b[c];
    #pragma unroll
    for (int q = 0; q < (2 * HG) / 4; ++q) {
        float4 a = yr[q], ww = wr[q];
        s += a.x * ww.x + a.y * ww.y + a.z * ww.z + a.w * ww.w;
    }
    out[i] = s;
}

extern "C" void kernel_launch(void* const* d_in, const int* in_sizes, int n_in,
                              void* d_out, int out_size, void* d_ws, size_t ws_size,
                              hipStream_t stream) {
    (void)in_sizes; (void)n_in; (void)out_size; (void)ws_size;

    const float* x     = (const float*)d_in[0];
    const int*   ei    = (const int*)d_in[1];      // [2, NE]: row=ei[e], col=ei[NE+e]
    const float* w1    = (const float*)d_in[2];    // [128,16]
    const float* b1    = (const float*)d_in[3];
    const float* w2    = (const float*)d_in[4];    // [16,32]
    const float* b2    = (const float*)d_in[5];
    const float* wih[4], *whh[4], *bih[4], *bhh[4];
    for (int l = 0; l < 4; ++l) {
        wih[l] = (const float*)d_in[6 + 4 * l + 0]; // [192, in]
        whh[l] = (const float*)d_in[6 + 4 * l + 1]; // [192, 64]
        bih[l] = (const float*)d_in[6 + 4 * l + 2]; // [192]
        bhh[l] = (const float*)d_in[6 + 4 * l + 3]; // [192]
    }
    const float* lin_w = (const float*)d_in[22];    // [10,128]
    const float* lin_b = (const float*)d_in[23];
    float* out = (float*)d_out;

    // workspace layout (floats)
    float* ws = (float*)d_ws;
    size_t off = 0;
    float* dinv = ws + off; off += NN;
    float* wpad = ws + off; off += 32 * 32;          // zero-padded W2
    float* xt   = ws + off; off += (size_t)NN * 32;  // staging (conv1: 16, conv2: 32)
    float* acc  = ws + off; off += (size_t)NN * 32;  // scatter accumulator / h
    float* gxf  = ws + off; off += (size_t)NN * G3;  // gate buffers, reused across layers
    float* gxb  = ws + off; off += (size_t)NN * G3;
    float* y0   = ws + off; off += (size_t)NN * 128;
    float* y1   = ws + off; off += (size_t)NN * 128;

    const int ET = NE + NN;
    const int TB = 256;

    // degrees -> D^{-1/2}
    hipMemsetAsync(dinv, 0, NN * sizeof(float), stream);
    k_degree<<<(ET + TB - 1) / TB, TB, 0, stream>>>(ei + NE, dinv, NE, NN);
    k_dinv<<<(NN + TB - 1) / TB, TB, 0, stream>>>(dinv, NN);
    k_pad_w2<<<(32 * 32 + TB - 1) / TB, TB, 0, stream>>>(w2, wpad);

    // ---- GCNConv 1: x@W1 -> scatter -> +b1, relu ----
    k_gemm_wmma<128, false, 1><<<dim3(NN / 16, 1), 32, 0, stream>>>(x, FIN, w1, 16, nullptr, xt, 16);
    hipMemsetAsync(acc, 0, (size_t)NN * 16 * sizeof(float), stream);
    k_scatter<16><<<(ET + TB - 1) / TB, TB, 0, stream>>>(ei, dinv, xt, acc, NE, NN);
    k_bias_relu<<<((NN * 16) + TB - 1) / TB, TB, 0, stream>>>(acc, b1, NN, 16);

    // ---- GCNConv 2: h1@W2pad (K padded to 32) -> scatter -> +b2, relu ----
    k_gemm_wmma<32, false, 2><<<dim3(NN / 16, 1), 32, 0, stream>>>(acc, 16, wpad, 32, nullptr, xt, 32);
    hipMemsetAsync(acc, 0, (size_t)NN * 32 * sizeof(float), stream);
    k_scatter<32><<<(ET + TB - 1) / TB, TB, 0, stream>>>(ei, dinv, xt, acc, NE, NN);
    k_bias_relu<<<((NN * 32) + TB - 1) / TB, TB, 0, stream>>>(acc, b2, NN, 32);

    // ---- GRU layer 0 (in=32): hoisted gate GEMMs + serial scan ----
    k_gemm_wmma<32, true, 12><<<dim3(NN / 16, 1), 32, 0, stream>>>(acc, 32, wih[0], 32, bih[0], gxf, G3);
    k_gemm_wmma<32, true, 12><<<dim3(NN / 16, 1), 32, 0, stream>>>(acc, 32, wih[1], 32, bih[1], gxb, G3);
    k_gru_scan<<<2, G3, 0, stream>>>(gxf, gxb, whh[0], bhh[0], whh[1], bhh[1], y0, NN);

    // ---- GRU layer 1 (in=128) ----
    k_gemm_wmma<128, true, 12><<<dim3(NN / 16, 1), 32, 0, stream>>>(y0, 128, wih[2], 128, bih[2], gxf, G3);
    k_gemm_wmma<128, true, 12><<<dim3(NN / 16, 1), 32, 0, stream>>>(y0, 128, wih[3], 128, bih[3], gxb, G3);
    k_gru_scan<<<2, G3, 0, stream>>>(gxf, gxb, whh[2], bhh[2], whh[3], bhh[3], y1, NN);

    // ---- classifier ----
    k_linear<<<((NN * NCLS) + TB - 1) / TB, TB, 0, stream>>>(y1, lin_w, lin_b, out, NN);
}
